// FactorioHGNN_14319420965080
// MI455X (gfx1250) — matrix operand order repaired and checked by reading, not compiled, not run on hardware
//
#include <hip/hip_runtime.h>
#include <hip/hip_bf16.h>
#include <stdint.h>

typedef _Float16 half_t;
typedef __attribute__((ext_vector_type(4)))  _Float16 v4h;
typedef __attribute__((ext_vector_type(8)))  _Float16 v8h;
typedef __attribute__((ext_vector_type(16))) _Float16 v16h;
typedef __attribute__((ext_vector_type(4)))  float    v4f;
typedef __attribute__((ext_vector_type(8)))  float    v8f;
typedef __attribute__((ext_vector_type(4)))  int      v4i;

#define N_NODES 8192
#define N_EDGES 4096
#define F_IN    512
#define HID     256

#define AS_GLOBAL __attribute__((address_space(1)))
#define AS_LOCAL  __attribute__((address_space(3)))

// Use gfx1250 async global->LDS copies when the toolchain exposes them.
#if defined(__has_builtin)
#  if __has_builtin(__builtin_amdgcn_global_load_async_to_lds_b128) && \
      __has_builtin(__builtin_amdgcn_s_wait_asynccnt)
#    define USE_ASYNC_LDS 1
#  endif
#endif
#ifndef USE_ASYNC_LDS
#  define USE_ASYNC_LDS 0
#endif

// ---------------- f32 -> f16 conversion (x4 vectorized) ----------------
__global__ void k_f32_to_f16(const float* __restrict__ src,
                             half_t* __restrict__ dst, int n4) {
  int i = blockIdx.x * blockDim.x + threadIdx.x;
  int stride = gridDim.x * blockDim.x;
  for (; i < n4; i += stride) {
    v4f v = ((const v4f*)src)[i];
    ((v4h*)dst)[i] = __builtin_convertvector(v, v4h);
  }
}

// ---------------- column sums of H -> 1/(d_e + eps) ----------------
__global__ void k_colsum_inv(const half_t* __restrict__ Hh,
                             float* __restrict__ dinv, int nrows, int ncols) {
  int e = blockIdx.x * blockDim.x + threadIdx.x;
  if (e >= ncols) return;
  float s = 0.f;
  for (int n = 0; n < nrows; ++n) s += (float)Hh[(size_t)n * ncols + e];
  dinv[e] = 1.0f / (s + 1e-6f);
}

// ---------------- WMMA GEMM: C(MxN) = A @ B, f16 in, f32 acc, f16 out ----
// TRANS_A=false: A is row-major MxK (lda=K)
// TRANS_A=true : A is stored KxM (lda=M), i.e. C = A_mem^T @ B  (used for H^T)
// EPI: 0 = none, 1 = multiply row m by dscale[m], 2 = relu
template <bool TRANS_A, int EPI>
__global__ __launch_bounds__(256) void k_gemm_wmma(
    const half_t* __restrict__ A, const half_t* __restrict__ B,
    half_t* __restrict__ C, const float* __restrict__ dscale,
    int M, int N, int K) {
  constexpr int BM = 128, BN = 128, BK = 32, LDT = BK + 8;  // 80B rows: 16B aligned
  __shared__ half_t a_lds[BM * LDT];  // a_lds[m][k]
  __shared__ half_t b_lds[BN * LDT];  // b_lds[n][k]

  const int tid  = threadIdx.x;
  const int lane = tid & 31;
  const int wave = tid >> 5;
  const int wm = (wave >> 2) * 64;   // wave M offset in block tile
  const int wn = (wave & 3) * 32;    // wave N offset
  const int m0 = blockIdx.y * BM;
  const int n0 = blockIdx.x * BN;
  const int lr = lane & 15;
  const int lh = lane >> 4;

  v8f acc[4][2] = {};

  for (int k0 = 0; k0 < K; k0 += BK) {
    // ---- stage A tile into a_lds[m][k] ----
    if constexpr (!TRANS_A) {
      // pure copy: 16B chunks contiguous in both global and LDS ->
      // gfx1250 async DMA path (no VGPR round trip), tracked by ASYNCcnt
      #pragma unroll
      for (int it = 0; it < 2; ++it) {
        int idx = tid + it * 256;          // 512 chunks of 8 halves
        int m   = idx >> 2;                // 4 chunks per row
        int kk  = (idx & 3) << 3;
        const half_t* gp = A + (size_t)(m0 + m) * K + (k0 + kk);
#if USE_ASYNC_LDS
        __builtin_amdgcn_global_load_async_to_lds_b128(
            (AS_GLOBAL v4i*)gp, (AS_LOCAL v4i*)&a_lds[m * LDT + kk], 0, 0);
#else
        *(v8h*)&a_lds[m * LDT + kk] = *(const v8h*)gp;
#endif
        if (k0 + BK < K)
          __builtin_prefetch(gp + BK, 0, 3);
      }
    } else {
      #pragma unroll
      for (int it = 0; it < 2; ++it) {
        int idx = tid + it * 256;
        int kk  = idx >> 4;                // 16 chunks per k-row
        int m   = (idx & 15) << 3;
        v8h v = *(const v8h*)(A + (size_t)(k0 + kk) * M + (m0 + m));
        #pragma unroll
        for (int j = 0; j < 8; ++j) a_lds[(m + j) * LDT + kk] = v[j];
        if (k0 + BK < K)
          __builtin_prefetch(A + (size_t)(k0 + BK + kk) * M + (m0 + m), 0, 3);
      }
    }
    // ---- stage B tile into b_lds[n][k] (transpose while staging) ----
    #pragma unroll
    for (int it = 0; it < 2; ++it) {
      int idx = tid + it * 256;
      int kk  = idx >> 4;
      int n   = (idx & 15) << 3;
      v8h v = *(const v8h*)(B + (size_t)(k0 + kk) * N + (n0 + n));
      #pragma unroll
      for (int j = 0; j < 8; ++j) b_lds[(n + j) * LDT + kk] = v[j];
      if (k0 + BK < K)
        __builtin_prefetch(B + (size_t)(k0 + BK + kk) * N + (n0 + n), 0, 3);
    }
#if USE_ASYNC_LDS
    if constexpr (!TRANS_A) __builtin_amdgcn_s_wait_asynccnt(0);
#endif
    __syncthreads();

    // ---- pack fragments, contiguous k order shared by A and B ----
    // B (32x16): lane -> n = lr, K = i + lh*16  => 16 consecutive k at [n][lh*16]
    v16h bf[2];
    #pragma unroll
    for (int nb = 0; nb < 2; ++nb) {
      const v8h* p = (const v8h*)&b_lds[(wn + nb * 16 + lr) * LDT + lh * 16];
      bf[nb] = __builtin_shufflevector(p[0], p[1],
               0,1,2,3,4,5,6,7,8,9,10,11,12,13,14,15);
    }
    // A (16x32): lane -> m = lr, K = i + lh*16  => 16 consecutive k at [m][lh*16]
    v16h af[4];
    #pragma unroll
    for (int mb = 0; mb < 4; ++mb) {
      const half_t* row = &a_lds[(wm + mb * 16 + lr) * LDT];
      v8h lo = *(const v8h*)(row + lh * 16);
      v8h hi = *(const v8h*)(row + lh * 16 + 8);
      af[mb] = __builtin_shufflevector(lo, hi,
               0,1,2,3,4,5,6,7,8,9,10,11,12,13,14,15);
    }

    #pragma unroll
    for (int mb = 0; mb < 4; ++mb) {
      #pragma unroll
      for (int nb = 0; nb < 2; ++nb) {
        acc[mb][nb] = __builtin_amdgcn_wmma_f32_16x16x32_f16(
            false, af[mb], false, bf[nb], (short)0, acc[mb][nb], false, false);
      }
    }
    __syncthreads();
  }

  // ---- epilogue: C/D layout: lane -> n = lr, row m = j + lh*8 ----
  #pragma unroll
  for (int mb = 0; mb < 4; ++mb) {
    #pragma unroll
    for (int nb = 0; nb < 2; ++nb) {
      int n = n0 + wn + nb * 16 + lr;
      #pragma unroll
      for (int j = 0; j < 8; ++j) {
        int m = m0 + wm + mb * 16 + lh * 8 + j;
        float v = acc[mb][nb][j];
        if (EPI == 1) v *= dscale[m];
        if (EPI == 2) v = v > 0.f ? v : 0.f;
        C[(size_t)m * N + n] = (half_t)v;
      }
    }
  }
}

// ---------------- column mean of h2 (8192 x 256) -> g ----------------
__global__ void k_zero(float* __restrict__ p, int n) {
  int i = blockIdx.x * blockDim.x + threadIdx.x;
  if (i < n) p[i] = 0.f;
}

__global__ void k_colmean(const half_t* __restrict__ h, float* __restrict__ g,
                          int rows_per_block, int F, float scale) {
  int f = threadIdx.x;            // blockDim.x == F == 256
  int n0 = blockIdx.x * rows_per_block;
  float s = 0.f;
  for (int r = 0; r < rows_per_block; ++r)
    s += (float)h[(size_t)(n0 + r) * F + f];
  atomicAdd(&g[f], s * scale);
}

// ---------------- tiny MLP heads (all f32) ----------------
__device__ void head_mlp(const float* gs, float* hid, int t,
                         const float* __restrict__ w1, const float* __restrict__ b1, int hd,
                         const float* __restrict__ w2, const float* __restrict__ b2, int od,
                         float* __restrict__ out) {
  if (t < hd) {
    float s = b1[t];
    for (int i = 0; i < HID; ++i) s = fmaf(gs[i], w1[i * hd + t], s);
    hid[t] = s > 0.f ? s : 0.f;
  }
  __syncthreads();
  if (t < od) {
    float s = b2[t];
    for (int j = 0; j < hd; ++j) s = fmaf(hid[j], w2[j * od + t], s);
    out[t] = s;
  }
  __syncthreads();
}

__global__ __launch_bounds__(320) void k_heads(
    const float* __restrict__ g,
    const float* aw1, const float* ab1, const float* aw2, const float* ab2,
    const float* iw1, const float* ib1, const float* iw2, const float* ib2,
    const float* hw1, const float* hb1, const float* hw2, const float* hb2,
    const float* rw1, const float* rb1, const float* rw2, const float* rb2,
    float* __restrict__ out) {
  __shared__ float gs[HID];
  __shared__ float hid[HID];
  int t = threadIdx.x;
  if (t < HID) gs[t] = g[t];
  __syncthreads();
  head_mlp(gs, hid, t, aw1, ab1, 128, aw2, ab2, 64,  out + 0);    // action
  head_mlp(gs, hid, t, iw1, ib1, 128, iw2, ib2, 256, out + 64);   // item
  head_mlp(gs, hid, t, hw1, hb1, 256, hw2, hb2, 289, out + 320);  // heatmap
  head_mlp(gs, hid, t, rw1, rb1, 128, rw2, rb2, 4,   out + 609);  // rotation
}

extern "C" void kernel_launch(void* const* d_in, const int* in_sizes, int n_in,
                              void* d_out, int out_size, void* d_ws, size_t ws_size,
                              hipStream_t stream) {
  (void)in_sizes; (void)n_in; (void)out_size; (void)ws_size;
  const float* x   = (const float*)d_in[0];
  const float* H   = (const float*)d_in[1];
  const float* W1  = (const float*)d_in[2];
  const float* W2  = (const float*)d_in[3];
  const float* aw1 = (const float*)d_in[4];
  const float* ab1 = (const float*)d_in[5];
  const float* aw2 = (const float*)d_in[6];
  const float* ab2 = (const float*)d_in[7];
  const float* iw1 = (const float*)d_in[8];
  const float* ib1 = (const float*)d_in[9];
  const float* iw2 = (const float*)d_in[10];
  const float* ib2 = (const float*)d_in[11];
  const float* hw1 = (const float*)d_in[12];
  const float* hb1 = (const float*)d_in[13];
  const float* hw2 = (const float*)d_in[14];
  const float* hb2 = (const float*)d_in[15];
  const float* rw1 = (const float*)d_in[16];
  const float* rb1 = (const float*)d_in[17];
  const float* rw2 = (const float*)d_in[18];
  const float* rb2 = (const float*)d_in[19];
  float* out = (float*)d_out;

  char* ws = (char*)d_ws;
  size_t off = 0;
  auto alloc = [&](size_t bytes) -> char* {
    char* p = ws + off;
    off += (bytes + 255) & ~(size_t)255;
    return p;
  };
  half_t* Hh   = (half_t*)alloc((size_t)N_NODES * N_EDGES * 2);  // 64 MB, L2-resident
  half_t* xh   = (half_t*)alloc((size_t)N_NODES * F_IN * 2);
  half_t* W1h  = (half_t*)alloc((size_t)F_IN * HID * 2);
  half_t* W2h  = (half_t*)alloc((size_t)HID * HID * 2);
  half_t* s1h  = (half_t*)alloc((size_t)N_EDGES * F_IN * 2);
  half_t* s2h  = (half_t*)alloc((size_t)N_NODES * F_IN * 2);
  half_t* h1h  = (half_t*)alloc((size_t)N_NODES * HID * 2);
  half_t* t1h  = (half_t*)alloc((size_t)N_EDGES * HID * 2);
  half_t* t2h  = (half_t*)alloc((size_t)N_NODES * HID * 2);
  half_t* h2h  = (half_t*)alloc((size_t)N_NODES * HID * 2);
  float*  dinv = (float*)alloc((size_t)N_EDGES * 4);
  float*  g    = (float*)alloc((size_t)HID * 4);

  // precision staging: H is exactly {0,1} in f16; x/W fit comfortably
  k_f32_to_f16<<<2048, 256, 0, stream>>>(H,  Hh,  N_NODES * N_EDGES / 4);
  k_f32_to_f16<<<512,  256, 0, stream>>>(x,  xh,  N_NODES * F_IN  / 4);
  k_f32_to_f16<<<64,   256, 0, stream>>>(W1, W1h, F_IN * HID / 4);
  k_f32_to_f16<<<32,   256, 0, stream>>>(W2, W2h, HID * HID / 4);
  k_colsum_inv<<<N_EDGES / 256, 256, 0, stream>>>(Hh, dinv, N_NODES, N_EDGES);

  dim3 blk(256);
  // layer 1:  s1 = diag(1/d) H^T x ;  s2 = H s1 ;  h1 = relu(s2 W1)
  k_gemm_wmma<true, 1><<<dim3(F_IN / 128, N_EDGES / 128), blk, 0, stream>>>(
      Hh, xh, s1h, dinv, N_EDGES, F_IN, N_NODES);
  k_gemm_wmma<false, 0><<<dim3(F_IN / 128, N_NODES / 128), blk, 0, stream>>>(
      Hh, s1h, s2h, nullptr, N_NODES, F_IN, N_EDGES);
  k_gemm_wmma<false, 2><<<dim3(HID / 128, N_NODES / 128), blk, 0, stream>>>(
      s2h, W1h, h1h, nullptr, N_NODES, HID, F_IN);
  // layer 2:  t1 = diag(1/d) H^T h1 ;  t2 = H t1 ;  h2 = relu(t2 W2)
  k_gemm_wmma<true, 1><<<dim3(HID / 128, N_EDGES / 128), blk, 0, stream>>>(
      Hh, h1h, t1h, dinv, N_EDGES, HID, N_NODES);
  k_gemm_wmma<false, 0><<<dim3(HID / 128, N_NODES / 128), blk, 0, stream>>>(
      Hh, t1h, t2h, nullptr, N_NODES, HID, N_EDGES);
  k_gemm_wmma<false, 2><<<dim3(HID / 128, N_NODES / 128), blk, 0, stream>>>(
      t2h, W2h, h2h, nullptr, N_NODES, HID, HID);

  // pooled embedding + heads
  k_zero<<<1, 256, 0, stream>>>(g, HID);
  k_colmean<<<32, 256, 0, stream>>>(h2h, g, N_NODES / 32, HID, 1.0f / N_NODES);
  k_heads<<<1, 320, 0, stream>>>(g, aw1, ab1, aw2, ab2, iw1, ib1, iw2, ib2,
                                 hw1, hb1, hw2, hb2, rw1, rb1, rw2, rb2, out);
}
